// SpatialAttention_49237505081438
// MI455X (gfx1250) — matrix operand order repaired and checked by reading, not compiled, not run on hardware
//
#include <hip/hip_runtime.h>
#include <hip/hip_bf16.h>

typedef __attribute__((ext_vector_type(16))) __bf16 v16bf;
typedef __attribute__((ext_vector_type(8)))  __bf16 v8bf;
typedef __attribute__((ext_vector_type(8)))  float  v8f;
typedef __attribute__((ext_vector_type(4)))  float  v4f;

#define B_SZ  2048
#define K_REG 49
#define D_SZ  2048
#define DC    128          // d-chunk staged in LDS per iteration
#define NPAD  64           // 49 padded to 64 (4 WMMA tiles each dim)

__global__ __launch_bounds__(256)
void spatial_attn_kernel(const float* __restrict__ V,
                         const float* __restrict__ h_t,
                         const float* __restrict__ s_t,
                         const float* __restrict__ W_v,
                         const float* __restrict__ W_g,
                         const float* __restrict__ W_s,
                         const float* __restrict__ w_h,
                         const float* __restrict__ w_h_s,
                         float* __restrict__ out) {
    __shared__ __align__(32) __bf16 Alds[NPAD][DC];   // V[b] chunk, bf16, zero-padded rows
    __shared__ __align__(32) __bf16 Blds[NPAD][DC];   // W_v chunk, bf16, zero-padded rows
    __shared__ float Clds[NPAD][66];                  // 64x64 f32 result (padded stride)
    __shared__ float Wgt[NPAD];                       // h_t[b] @ W_g^T
    __shared__ float Wsr[NPAD];                       // s_t[b] @ W_s^T
    __shared__ float wh_l[NPAD];
    __shared__ float whs_l[NPAD];
    __shared__ float alpha[NPAD];                     // logits -> softmax probs

    const int tid  = threadIdx.x;
    const int lane = tid & 31;
    const int wave = tid >> 5;
    const int b    = blockIdx.x;

    const float* Vb = V   + (size_t)b * K_REG * D_SZ;
    const float* hb = h_t + (size_t)b * D_SZ;
    const float* sb = s_t + (size_t)b * D_SZ;

    if (tid < K_REG) { wh_l[tid] = w_h[tid]; whs_l[tid] = w_h_s[tid]; }

    // ---- Phase 1: small matvecs, one k per wave, b128 loads, wave32 shfl reduce ----
    for (int k = wave; k < K_REG; k += 8) {
        const float* wg = W_g + (size_t)k * D_SZ;
        const float* ws = W_s + (size_t)k * D_SZ;
        v4f agv = {}, asv = {};
        for (int d = lane << 2; d < D_SZ; d += 128) {
            const v4f hv  = *(const v4f*)&hb[d];
            const v4f sv  = *(const v4f*)&sb[d];
            const v4f wgv = *(const v4f*)&wg[d];
            const v4f wsv = *(const v4f*)&ws[d];
            agv += hv * wgv;
            asv += sv * wsv;
        }
        float ag = agv[0] + agv[1] + agv[2] + agv[3];
        float as = asv[0] + asv[1] + asv[2] + asv[3];
        for (int off = 16; off; off >>= 1) {
            ag += __shfl_xor(ag, off, 32);
            as += __shfl_xor(as, off, 32);
        }
        if (lane == 0) { Wgt[k] = ag; Wsr[k] = as; }
    }

    // ---- Phase 2: 64x64x2048 GEMM via v_wmma_f32_16x16x32_bf16 ----
    const int mtile = wave >> 1;
    const int nt0   = (wave & 1) * 2;
    const int arow  = mtile * 16 + (lane & 15);
    const int abase = (lane < 16) ? 0 : 16;    // A lanes split K: {0..15} / {16..31}
    const int brow0 = nt0 * 16 + (lane & 15);  // B lane n = column n = W_v row n (K = contiguous d)
    const int brow1 = brow0 + 16;
    const int bbase = (lane < 16) ? 0 : 16;    // lanes 0-15: K 0-15, lanes 16-31: K 16-31

    v8f c0 = {}, c1 = {};

    for (int c = 0; c < D_SZ; c += DC) {
        __syncthreads();
        // vectorized f32 -> bf16 staging: 8 floats (2x b128) -> 8 bf16 (1x b128 DS store)
        #pragma unroll
        for (int it = 0; it < (NPAD * DC / 8) / 256; ++it) {
            const int g   = tid + it * 256;
            const int r   = g >> 4;            // 16 groups of 8 per 128-wide row
            const int col = (g & 15) << 3;
            v8bf av = {}, bv = {};
            if (r < K_REG) {
                const v4f* pa = (const v4f*)&Vb [(size_t)r * D_SZ + c + col];
                const v4f* pb = (const v4f*)&W_v[(size_t)r * D_SZ + c + col];
                const v4f a0 = pa[0], a1 = pa[1];
                const v4f b0 = pb[0], b1 = pb[1];
                #pragma unroll
                for (int j = 0; j < 4; ++j) {
                    av[j]     = (__bf16)a0[j];
                    av[4 + j] = (__bf16)a1[j];
                    bv[j]     = (__bf16)b0[j];
                    bv[4 + j] = (__bf16)b1[j];
                }
            }
            *(v8bf*)&Alds[r][col] = av;
            *(v8bf*)&Blds[r][col] = bv;
        }
        __syncthreads();

        #pragma unroll
        for (int d0 = 0; d0 < DC; d0 += 32) {
            v16bf a  = *(const v16bf*)&Alds[arow][d0 + abase];
            v16bf b0 = *(const v16bf*)&Blds[brow0][d0 + bbase];
            v16bf b1 = *(const v16bf*)&Blds[brow1][d0 + bbase];
            c0 = __builtin_amdgcn_wmma_f32_16x16x32_bf16(false, a, false, b0,
                                                         (short)0, c0, false, false);
            c1 = __builtin_amdgcn_wmma_f32_16x16x32_bf16(false, a, false, b1,
                                                         (short)0, c1, false, false);
        }
    }

    // scatter accumulators to LDS: VGPR i -> row (i + 8*(lane>=16)), col = lane&15
    {
        const int rbase = mtile * 16 + ((lane >> 4) << 3);
        const int col   = lane & 15;
        #pragma unroll
        for (int i = 0; i < 8; ++i) {
            Clds[rbase + i][nt0 * 16 + col]      = c0[i];
            Clds[rbase + i][nt0 * 16 + 16 + col] = c1[i];
        }
    }
    __syncthreads();

    // ---- Phase 3: logits (tanh + dot), softmax over 50 ----
    float my_logit = 0.f;
    if (tid < K_REG + 1) {
        const bool sent = (tid == K_REG);
        float acc = 0.f;
        for (int k = 0; k < K_REG; ++k) {
            const float base = sent ? Wsr[k] : Clds[tid][k];
            const float t    = tanhf(base + Wgt[k]);
            acc += t * (sent ? whs_l[k] : wh_l[k]);
        }
        my_logit = acc;
        alpha[tid] = acc;          // raw logits first
    }
    __syncthreads();

    float mx = -1e30f;
    for (int i = 0; i < K_REG + 1; ++i) mx = fmaxf(mx, alpha[i]);
    float ssum = 0.f;
    for (int i = 0; i < K_REG + 1; ++i) ssum += expf(alpha[i] - mx);
    const float inv = 1.f / ssum;
    __syncthreads();
    if (tid < K_REG + 1) alpha[tid] = expf(my_logit - mx) * inv;
    __syncthreads();

    // ---- Phase 4: c_t = alpha^T V[b] (L2-hot re-read, b128), blend, store ----
    {
        const float beta = alpha[K_REG];
        v4f acc0 = {}, acc1 = {};
        for (int l = 0; l < K_REG; ++l) {
            const float a  = alpha[l];
            const v4f* pv = (const v4f*)&Vb[(size_t)l * D_SZ];
            acc0 += a * pv[tid];
            acc1 += a * pv[tid + 256];
        }
        const v4f* ph = (const v4f*)hb;
        const v4f* ps = (const v4f*)sb;
        v4f* po = (v4f*)(out + (size_t)b * D_SZ);
        const float ob = 1.f - beta;
        po[tid]       = beta * ps[tid]       + ob * acc0 + ph[tid];
        po[tid + 256] = beta * ps[tid + 256] + ob * acc1 + ph[tid + 256];
    }
}

extern "C" void kernel_launch(void* const* d_in, const int* in_sizes, int n_in,
                              void* d_out, int out_size, void* d_ws, size_t ws_size,
                              hipStream_t stream) {
    const float* V     = (const float*)d_in[0];
    const float* h_t   = (const float*)d_in[1];
    const float* s_t   = (const float*)d_in[2];
    const float* W_v   = (const float*)d_in[3];
    const float* W_g   = (const float*)d_in[4];
    const float* W_s   = (const float*)d_in[5];
    const float* w_h   = (const float*)d_in[6];
    const float* w_h_s = (const float*)d_in[7];
    float* out = (float*)d_out;

    spatial_attn_kernel<<<dim3(B_SZ), dim3(256), 0, stream>>>(
        V, h_t, s_t, W_v, W_g, W_s, w_h, w_h_s, out);
}
